// DecoderBlock_39273180955253
// MI455X (gfx1250) — matrix operand order repaired and checked by reading, not compiled, not run on hardware
//
#include <hip/hip_runtime.h>
#include <hip/hip_bf16.h>

// ---------------------------------------------------------------------------
// MLA decoder block for gfx1250 (MI455X). All GEMMs via v_wmma_f32_16x16x32_f16.
// LDS staging uses CDNA5 async memory->LDS DMA (global_load_async_to_lds_b128,
// tracked by ASYNCcnt) instead of a synchronous VGPR round trip.
// ---------------------------------------------------------------------------

typedef __attribute__((ext_vector_type(16))) _Float16 v16h;
typedef __attribute__((ext_vector_type(8)))  _Float16 v8h;
typedef __attribute__((ext_vector_type(8)))  float    v8f;

#define NB 4
#define NS 1024
#define ND 2048
#define NH 16
#define DN 128
#define DROT 64
#define DQK 192     // DN + DR
#define NQR 1024
#define NKVR 512
#define NFF 5632
#define NTOK (NB*NS)   // 4096

// ---- CDNA5 async copy helpers (ISA: GLOBAL_LOAD_ASYNC_TO_LDS_B128 / ASYNCcnt)
__device__ __forceinline__ void async_copy_b128(const void* gptr, void* lptr)
{
    const unsigned lds = (unsigned)(unsigned long long)lptr;   // LDS offset
    asm volatile("global_load_async_to_lds_b128 %0, %1, off"
                 :: "v"(lds), "v"(gptr) : "memory");
}
__device__ __forceinline__ void wait_async_zero()
{
    asm volatile("s_wait_asynccnt 0x0" ::: "memory");
}

// ===================== weight transpose+convert (once) =====================
// out[N][K] (f16) = in[K][N] (f32) ; tiled via LDS to keep both sides coalesced
__global__ __launch_bounds__(256)
void transpose_convert(const float* __restrict__ in, _Float16* __restrict__ out,
                       int K, int N)
{
    __shared__ float tile[32][33];
    const int k0 = blockIdx.y * 32, n0 = blockIdx.x * 32;
    const int tx = threadIdx.x & 31, ty = threadIdx.x >> 5;   // 32 x 8
#pragma unroll
    for (int r = ty; r < 32; r += 8) {
        const int k = k0 + r, n = n0 + tx;
        tile[r][tx] = (k < K && n < N) ? in[(size_t)k * N + n] : 0.0f;
    }
    __syncthreads();
#pragma unroll
    for (int r = ty; r < 32; r += 8) {
        const int n = n0 + r, k = k0 + tx;
        if (n < N && k < K) out[(size_t)n * K + k] = (_Float16)tile[tx][r];
    }
}

// ============================ generic WMMA GEMM ============================
// C[M,N](f32) = A[M,K](f16 row major) * BT[N,K](f16, pre-transposed weights)
//               (+ resid[M,N] if non-null)
// Tile 128x128x32, 256 threads = 8 waves in 2x4, each wave owns 64x32.
#define TM 128
#define TN 128
#define TK 32

__global__ __launch_bounds__(256)
void gemm_wmma_f16(const _Float16* __restrict__ A, const _Float16* __restrict__ BT,
                   const float* __restrict__ resid, float* __restrict__ outF,
                   int M, int N, int K)
{
    __shared__ __align__(16) _Float16 sA [TM * TK];
    __shared__ __align__(16) _Float16 sBt[TN * TK];   // [col][k]

    const int tid  = threadIdx.x;
    const int lane = tid & 31;
    const int wave = tid >> 5;
    const int l16  = lane & 15;
    const int hi   = lane >> 4;
    const int wm   = (wave >> 2) * 64;   // wave row offset inside tile
    const int wn   = (wave & 3) * 32;    // wave col offset inside tile
    const int m0   = blockIdx.y * TM;
    const int n0   = blockIdx.x * TN;

    v8f acc[4][2];
#pragma unroll
    for (int mt = 0; mt < 4; ++mt)
#pragma unroll
        for (int nt = 0; nt < 2; ++nt)
#pragma unroll
            for (int j = 0; j < 8; ++j) acc[mt][nt][j] = 0.0f;

    for (int k0 = 0; k0 < K; k0 += TK) {
        // ---- async DMA stage: A tile 128x32, BT tile 128x32 (b128 chunks) ----
#pragma unroll
        for (int i = tid; i < (TM * TK) / 8; i += 256) {
            const int r = i >> 2;
            const int c = (i & 3) << 3;
            async_copy_b128(A + (size_t)(m0 + r) * K + k0 + c, &sA[r * TK + c]);
        }
#pragma unroll
        for (int i = tid; i < (TN * TK) / 8; i += 256) {
            const int r = i >> 2;
            const int c = (i & 3) << 3;
            int gr = n0 + r;
            if (gr >= N) gr = N - 1;   // clamp; those cols are discarded in epilogue
            async_copy_b128(BT + (size_t)gr * K + k0 + c, &sBt[r * TK + c]);
        }
        // L2 prefetch of next BT tile
        if (k0 + TK < K) {
            __builtin_prefetch(BT + (size_t)(n0 + (tid >> 1)) * K + k0 + TK + (tid & 1) * 16, 0, 1);
        }
        wait_async_zero();
        __syncthreads();

        // ---- B fragments: lane col = wn+nt*16+l16, K segs at hi*8 and 16+hi*8 ----
        v16h bf[2];
#pragma unroll
        for (int nt = 0; nt < 2; ++nt) {
            const _Float16* br = &sBt[(wn + nt * 16 + l16) * TK];
            const v8h lo = *(const v8h*)(br + hi * 8);
            const v8h hh = *(const v8h*)(br + 16 + hi * 8);
#pragma unroll
            for (int j = 0; j < 8; ++j) { bf[nt][j] = lo[j]; bf[nt][8 + j] = hh[j]; }
        }
        // ---- A fragments + WMMA ----
#pragma unroll
        for (int mt = 0; mt < 4; ++mt) {
            const _Float16* ar = &sA[(wm + mt * 16 + l16) * TK];
            const v8h lo = *(const v8h*)(ar + hi * 8);
            const v8h hh = *(const v8h*)(ar + 16 + hi * 8);
            v16h af;
#pragma unroll
            for (int j = 0; j < 8; ++j) { af[j] = lo[j]; af[8 + j] = hh[j]; }
#pragma unroll
            for (int nt = 0; nt < 2; ++nt) {
                acc[mt][nt] = __builtin_amdgcn_wmma_f32_16x16x32_f16(
                    false, af, false, bf[nt], (short)0, acc[mt][nt], false, false);
            }
        }
        __syncthreads();
    }

    // ---- epilogue: C layout row = r + 8*hi, col = l16 ----
#pragma unroll
    for (int mt = 0; mt < 4; ++mt) {
#pragma unroll
        for (int nt = 0; nt < 2; ++nt) {
            const int gn = n0 + wn + nt * 16 + l16;
            if (gn < N) {
#pragma unroll
                for (int r = 0; r < 8; ++r) {
                    const int gm = m0 + wm + mt * 16 + r + 8 * hi;
                    if (gm < M) {
                        float v = acc[mt][nt][r];
                        if (resid) v += resid[(size_t)gm * N + gn];
                        outF[(size_t)gm * N + gn] = v;
                    }
                }
            }
        }
    }
}

// ======================= flash attention (WMMA, causal) =====================
// qh : [B,H,S,192] f16   kh : [B,H,S,192] f16   vT : [B,H,128,S] f16
// o16: [B,S,H*128] f16
// grid (S/64, H, B), 128 threads = 4 waves; wave w handles q rows [64*qt+16w,+16)
__global__ __launch_bounds__(128)
void attn_wmma(const _Float16* __restrict__ qh, const _Float16* __restrict__ kh,
               const _Float16* __restrict__ vT, _Float16* __restrict__ o16)
{
    const int b = blockIdx.z, h = blockIdx.y, qt = blockIdx.x;
    const int tid  = threadIdx.x;
    const int lane = tid & 31;
    const int wave = tid >> 5;
    const int l16  = lane & 15;
    const int hi   = lane >> 4;
    const int q0   = qt * 64;

    __shared__ __align__(16) _Float16 sK [64 * DQK];  // [key][d] stride 192
    __shared__ __align__(16) _Float16 sVT[DN * 64];   // [d][key] stride 64
    __shared__ __align__(16) _Float16 sP [4][16 * 64];

    // preload the wave's Q A-fragments straight from global (6 K-chunks of 32)
    v16h qa[6];
    {
        const _Float16* qrow =
            qh + (((size_t)b * NH + h) * NS + q0 + wave * 16 + l16) * DQK;
#pragma unroll
        for (int c = 0; c < 6; ++c) {
            const v8h lo = *(const v8h*)(qrow + c * 32 + hi * 8);
            const v8h hh = *(const v8h*)(qrow + c * 32 + 16 + hi * 8);
#pragma unroll
            for (int j = 0; j < 8; ++j) { qa[c][j] = lo[j]; qa[c][8 + j] = hh[j]; }
        }
    }

    v8f oacc[8];
#pragma unroll
    for (int nv = 0; nv < 8; ++nv)
#pragma unroll
        for (int j = 0; j < 8; ++j) oacc[nv][j] = 0.0f;
    float rmax[8], rsum[8];
#pragma unroll
    for (int r = 0; r < 8; ++r) { rmax[r] = -3.0e38f; rsum[r] = 0.0f; }

    const _Float16* kbase  = kh + ((size_t)b * NH + h) * (size_t)NS * DQK;
    const _Float16* vtbase = vT + ((size_t)b * NH + h) * (size_t)DN * NS;
    const float scale = 0.07216878364870322f;  // 1/sqrt(192)

    for (int kt = 0; kt <= qt; ++kt) {
        // async DMA stage: K tile 64x192 (row major) and V^T tile 128x64
        for (int i = tid; i < (64 * DQK) / 8; i += 128) {      // 1536 chunks
            const int r = i / 24;
            const int c = (i % 24) << 3;
            async_copy_b128(kbase + (size_t)(kt * 64 + r) * DQK + c,
                            &sK[r * DQK + c]);
        }
#pragma unroll
        for (int i = tid; i < (DN * 64) / 8; i += 128) {       // 1024 chunks
            const int r = i >> 3, c = (i & 7) << 3;
            async_copy_b128(vtbase + (size_t)r * NS + kt * 64 + c,
                            &sVT[r * 64 + c]);
        }
        wait_async_zero();
        __syncthreads();

        // scores S = Q * K^T  (16x64 per wave, feature dim 192)
        v8f sacc[4];
#pragma unroll
        for (int nt = 0; nt < 4; ++nt)
#pragma unroll
            for (int j = 0; j < 8; ++j) sacc[nt][j] = 0.0f;
#pragma unroll
        for (int c = 0; c < 6; ++c) {
#pragma unroll
            for (int nt = 0; nt < 4; ++nt) {
                const _Float16* br = &sK[(nt * 16 + l16) * DQK + c * 32];
                const v8h lo = *(const v8h*)(br + hi * 8);
                const v8h hh = *(const v8h*)(br + 16 + hi * 8);
                v16h bfr;
#pragma unroll
                for (int j = 0; j < 8; ++j) { bfr[j] = lo[j]; bfr[8 + j] = hh[j]; }
                sacc[nt] = __builtin_amdgcn_wmma_f32_16x16x32_f16(
                    false, qa[c], false, bfr, (short)0, sacc[nt], false, false);
            }
        }

        // online softmax (rows r+8*hi; reductions across the 16-lane half-wave)
#pragma unroll
        for (int r = 0; r < 8; ++r) {
            const int grow = q0 + wave * 16 + r + 8 * hi;
            float tm = -3.0e38f;
#pragma unroll
            for (int nt = 0; nt < 4; ++nt) {
                const int col = kt * 64 + nt * 16 + l16;
                float s = sacc[nt][r] * scale;
                if (col > grow) s = -3.0e38f;
                sacc[nt][r] = s;
                tm = fmaxf(tm, s);
            }
            tm = fmaxf(tm, __shfl_xor(tm, 1));
            tm = fmaxf(tm, __shfl_xor(tm, 2));
            tm = fmaxf(tm, __shfl_xor(tm, 4));
            tm = fmaxf(tm, __shfl_xor(tm, 8));
            const float nm   = fmaxf(rmax[r], tm);
            const float corr = __expf(rmax[r] - nm);
            rmax[r] = nm;
            float ps = 0.0f;
#pragma unroll
            for (int nt = 0; nt < 4; ++nt) {
                const float p = __expf(sacc[nt][r] - nm);
                sacc[nt][r] = p;
                ps += p;
            }
            ps += __shfl_xor(ps, 1);
            ps += __shfl_xor(ps, 2);
            ps += __shfl_xor(ps, 4);
            ps += __shfl_xor(ps, 8);
            rsum[r] = rsum[r] * corr + ps;
#pragma unroll
            for (int nv = 0; nv < 8; ++nv) oacc[nv][r] *= corr;
        }

        // scatter P (C layout) to this wave's LDS region, reload in A layout
#pragma unroll
        for (int r = 0; r < 8; ++r)
#pragma unroll
            for (int nt = 0; nt < 4; ++nt)
                sP[wave][(r + 8 * hi) * 64 + nt * 16 + l16] = (_Float16)sacc[nt][r];

        // O += P * V   (K-dim 64 = 2 chunks, 8 n-tiles of 16)
#pragma unroll
        for (int c = 0; c < 2; ++c) {
            const _Float16* pr = &sP[wave][l16 * 64 + c * 32];
            const v8h plo = *(const v8h*)(pr + hi * 8);
            const v8h phh = *(const v8h*)(pr + 16 + hi * 8);
            v16h pa;
#pragma unroll
            for (int j = 0; j < 8; ++j) { pa[j] = plo[j]; pa[8 + j] = phh[j]; }
#pragma unroll
            for (int nv = 0; nv < 8; ++nv) {
                const _Float16* br = &sVT[(nv * 16 + l16) * 64 + c * 32];
                const v8h lo = *(const v8h*)(br + hi * 8);
                const v8h hh = *(const v8h*)(br + 16 + hi * 8);
                v16h bfr;
#pragma unroll
                for (int j = 0; j < 8; ++j) { bfr[j] = lo[j]; bfr[8 + j] = hh[j]; }
                oacc[nv] = __builtin_amdgcn_wmma_f32_16x16x32_f16(
                    false, pa, false, bfr, (short)0, oacc[nv], false, false);
            }
        }
        __syncthreads();
    }

    // write O (f16) into [B,S,H*128]
#pragma unroll
    for (int nv = 0; nv < 8; ++nv) {
        const int coln = h * DN + nv * 16 + l16;
#pragma unroll
        for (int r = 0; r < 8; ++r) {
            const int grow = q0 + wave * 16 + r + 8 * hi;
            o16[((size_t)b * NS + grow) * (size_t)ND + coln] =
                (_Float16)(oacc[nv][r] / rsum[r]);
        }
    }
}

// ============================ elementwise kernels ==========================
__global__ __launch_bounds__(256)
void rmsnorm_f32_to_f16(const float* __restrict__ in, const float* __restrict__ w,
                        _Float16* __restrict__ out, int ncols, int instride, int outstride)
{
    const int row = blockIdx.x;
    const float* x = in + (size_t)row * instride;
    __shared__ float red[8];
    float ss = 0.0f;
    for (int c = threadIdx.x; c < ncols; c += 256) { const float v = x[c]; ss += v * v; }
#pragma unroll
    for (int off = 1; off < 32; off <<= 1) ss += __shfl_xor(ss, off);
    if ((threadIdx.x & 31) == 0) red[threadIdx.x >> 5] = ss;
    __syncthreads();
    if (threadIdx.x == 0) {
        float t = 0.0f;
#pragma unroll
        for (int i = 0; i < 8; ++i) t += red[i];
        red[0] = t;
    }
    __syncthreads();
    const float inv = rsqrtf(red[0] / (float)ncols + 1e-6f);
    for (int c = threadIdx.x; c < ncols; c += 256)
        out[(size_t)row * outstride + c] = (_Float16)(x[c] * inv * w[c]);
}

// q32 [tok,3072]=[b,s,h,192] -> qh16 [B,H,S,192] with RoPE on last 64 dims
__global__ __launch_bounds__(256)
void rope_pack_q(const float* __restrict__ q32, _Float16* __restrict__ qh)
{
    const int idx = blockIdx.x * blockDim.x + threadIdx.x;   // tok*16 + h
    if (idx >= NTOK * NH) return;
    const int tok = idx >> 4, h = idx & 15;
    const int b = tok >> 10, s = tok & 1023;
    const float* q = q32 + (size_t)tok * (NH * DQK) + h * DQK;
    _Float16* o = qh + (((size_t)b * NH + h) * NS + s) * DQK;
#pragma unroll 8
    for (int d = 0; d < DN; ++d) o[d] = (_Float16)q[d];
    for (int i = 0; i < 32; ++i) {
        const float ang = (float)s * __expf(-((float)(2 * i) / 64.0f) * 9.210340371976184f);
        const float cs = __cosf(ang), sn = __sinf(ang);
        const float x1 = q[DN + i], x2 = q[DN + 32 + i];
        o[DN + i]      = (_Float16)(x1 * cs - x2 * sn);
        o[DN + 32 + i] = (_Float16)(x2 * cs + x1 * sn);
    }
}

// kv32 [tok,4096]=[tok,h,256], ckv32 [tok,576] (k_rope = cols 512..575)
// -> kh [B,H,S,192] (row major), vT [B,H,128,S] (transposed)
__global__ __launch_bounds__(256)
void rope_pack_k(const float* __restrict__ kv32, const float* __restrict__ ckv32,
                 _Float16* __restrict__ kh, _Float16* __restrict__ vT)
{
    const int idx = blockIdx.x * blockDim.x + threadIdx.x;   // tok*16 + h
    if (idx >= NTOK * NH) return;
    const int tok = idx >> 4, h = idx & 15;
    const int b = tok >> 10, s = tok & 1023;
    const float* kvp = kv32 + (size_t)tok * (NH * 2 * DN) + h * (2 * DN);
    const float* kr  = ckv32 + (size_t)tok * (NKVR + DROT) + NKVR;
    _Float16* ko = kh + (((size_t)b * NH + h) * NS + s) * DQK;
    _Float16* vo = vT + ((size_t)b * NH + h) * (size_t)DN * NS + s;
#pragma unroll 8
    for (int d = 0; d < DN; ++d) {
        ko[d] = (_Float16)kvp[d];
        vo[(size_t)d * NS] = (_Float16)kvp[DN + d];
    }
    for (int i = 0; i < 32; ++i) {
        const float ang = (float)s * __expf(-((float)(2 * i) / 64.0f) * 9.210340371976184f);
        const float cs = __cosf(ang), sn = __sinf(ang);
        const float k1 = kr[i], k2 = kr[32 + i];
        ko[DN + i]      = (_Float16)(k1 * cs - k2 * sn);
        ko[DN + 32 + i] = (_Float16)(k2 * cs + k1 * sn);
    }
}

__global__ __launch_bounds__(256)
void silu_mul(const float* __restrict__ g, const float* __restrict__ u,
              _Float16* __restrict__ out, int n)
{
    const int i = blockIdx.x * blockDim.x + threadIdx.x;
    if (i < n) {
        const float x = g[i];
        const float s = x / (1.0f + __expf(-x));
        out[i] = (_Float16)(s * u[i]);
    }
}

// ============================== host launcher ==============================
extern "C" void kernel_launch(void* const* d_in, const int* in_sizes, int n_in,
                              void* d_out, int out_size, void* d_ws, size_t ws_size,
                              hipStream_t stream)
{
    (void)in_sizes; (void)n_in; (void)out_size; (void)ws_size;

    const float* x           = (const float*)d_in[0];
    const float* attn_norm_w = (const float*)d_in[1];
    const float* wq_down     = (const float*)d_in[2];
    const float* q_norm_w    = (const float*)d_in[3];
    const float* wq_up       = (const float*)d_in[4];
    const float* wkv_down    = (const float*)d_in[5];
    const float* kv_norm_w   = (const float*)d_in[6];
    const float* wkv_up      = (const float*)d_in[7];
    const float* wo          = (const float*)d_in[8];
    const float* ffn_norm_w  = (const float*)d_in[9];
    const float* w_gate      = (const float*)d_in[10];
    const float* w_up        = (const float*)d_in[11];
    const float* w_down      = (const float*)d_in[12];
    float* out = (float*)d_out;

    // ---- workspace arena ----
    char* ws = (char*)d_ws;
    // transposed f16 weights (persist entire launch)
    constexpr size_t OFF_WQD = 0;                    // [1024][2048]  4 MB
    constexpr size_t OFF_WQU = OFF_WQD + 4194304;    // [3072][1024]  6 MB
    constexpr size_t OFF_WKD = OFF_WQU + 6291456;    // [576][2048]   2.25 MB
    constexpr size_t OFF_WKU = OFF_WKD + 2359296;    // [4096][512]   4 MB
    constexpr size_t OFF_WO  = OFF_WKU + 4194304;    // [2048][2048]  8 MB
    constexpr size_t OFF_WG  = OFF_WO  + 8388608;    // [5632][2048]  22 MB
    constexpr size_t OFF_WU  = OFF_WG  + 23068672;   // [5632][2048]  22 MB
    constexpr size_t OFF_WD  = OFF_WU  + 23068672;   // [2048][5632]  22 MB
    constexpr size_t OFF_XA  = OFF_WD  + 23068672;   // xa16 16 MB
    constexpr size_t OFF_R1  = OFF_XA  + 16777216;   // q32(48MB) -> gate32(88MB)
    constexpr size_t OFF_R2  = OFF_R1  + 92274688;   // kv32(64MB) -> up32(88MB)
    constexpr size_t OFF_R3  = OFF_R2  + 92274688;   // ckv/cq -> a16(44MB)
    constexpr size_t OFF_R4  = OFF_R3  + 46137344;   // qh/kh/vT -> h32/hf16
    constexpr size_t OFF_R5  = OFF_R4  + 67108864;   // o16 16 MB

    _Float16* wqdT = (_Float16*)(ws + OFF_WQD);
    _Float16* wquT = (_Float16*)(ws + OFF_WQU);
    _Float16* wkdT = (_Float16*)(ws + OFF_WKD);
    _Float16* wkuT = (_Float16*)(ws + OFF_WKU);
    _Float16* woT  = (_Float16*)(ws + OFF_WO);
    _Float16* wgT  = (_Float16*)(ws + OFF_WG);
    _Float16* wuT  = (_Float16*)(ws + OFF_WU);
    _Float16* wdT  = (_Float16*)(ws + OFF_WD);

    _Float16* xa16   = (_Float16*)(ws + OFF_XA);
    float*    q32    = (float*)(ws + OFF_R1);
    float*    gate32 = (float*)(ws + OFF_R1);
    float*    kv32   = (float*)(ws + OFF_R2);
    float*    up32   = (float*)(ws + OFF_R2);
    float*    ckv32  = (float*)(ws + OFF_R3);
    float*    cq32   = (float*)(ws + OFF_R3 + 9437184);
    _Float16* cq16   = (_Float16*)(ws + OFF_R3 + 9437184 + 16777216);
    _Float16* ckvn16 = (_Float16*)(ws + OFF_R3 + 9437184 + 16777216 + 8388608);
    _Float16* a16    = (_Float16*)(ws + OFF_R3);               // reuse after silu inputs done
    _Float16* qh16   = (_Float16*)(ws + OFF_R4);
    _Float16* kh16   = (_Float16*)(ws + OFF_R4 + 25165824);
    _Float16* vT16   = (_Float16*)(ws + OFF_R4 + 50331648);
    float*    h32    = (float*)(ws + OFF_R4);                  // reuse after attention
    _Float16* hf16   = (_Float16*)(ws + OFF_R4 + 33554432);
    _Float16* o16    = (_Float16*)(ws + OFF_R5);

    const dim3 blk256(256), blk128(128);
    auto ggrid = [](int N, int M) { return dim3((N + TN - 1) / TN, (M + TM - 1) / TM); };
    auto tgrid = [](int K, int N) { return dim3((N + 31) / 32, (K + 31) / 32); };

    // 0. transpose + f16-convert all weights
    transpose_convert<<<tgrid(ND, NQR),         blk256, 0, stream>>>(wq_down,  wqdT, ND, NQR);
    transpose_convert<<<tgrid(NQR, NH * DQK),   blk256, 0, stream>>>(wq_up,    wquT, NQR, NH * DQK);
    transpose_convert<<<tgrid(ND, NKVR + DROT), blk256, 0, stream>>>(wkv_down, wkdT, ND, NKVR + DROT);
    transpose_convert<<<tgrid(NKVR, NH * 2 * DN), blk256, 0, stream>>>(wkv_up, wkuT, NKVR, NH * 2 * DN);
    transpose_convert<<<tgrid(ND, ND),          blk256, 0, stream>>>(wo,       woT,  ND, ND);
    transpose_convert<<<tgrid(ND, NFF),         blk256, 0, stream>>>(w_gate,   wgT,  ND, NFF);
    transpose_convert<<<tgrid(ND, NFF),         blk256, 0, stream>>>(w_up,     wuT,  ND, NFF);
    transpose_convert<<<tgrid(NFF, ND),         blk256, 0, stream>>>(w_down,   wdT,  NFF, ND);

    // 1. xa = rmsnorm(x) -> f16
    rmsnorm_f32_to_f16<<<NTOK, blk256, 0, stream>>>(x, attn_norm_w, xa16, ND, ND, ND);
    // 2. cq = xa @ wq_down
    gemm_wmma_f16<<<ggrid(NQR, NTOK), blk256, 0, stream>>>(xa16, wqdT, nullptr, cq32, NTOK, NQR, ND);
    // 3. cqn = rmsnorm(cq) -> f16
    rmsnorm_f32_to_f16<<<NTOK, blk256, 0, stream>>>(cq32, q_norm_w, cq16, NQR, NQR, NQR);
    // 4. q = cqn @ wq_up
    gemm_wmma_f16<<<ggrid(NH * DQK, NTOK), blk256, 0, stream>>>(cq16, wquT, nullptr, q32, NTOK, NH * DQK, NQR);
    // 5. ckv = xa @ wkv_down   (N = 576)
    gemm_wmma_f16<<<ggrid(NKVR + DROT, NTOK), blk256, 0, stream>>>(xa16, wkdT, nullptr, ckv32, NTOK, NKVR + DROT, ND);
    // 6. c_kv norm -> f16 (512 of 576 cols)
    rmsnorm_f32_to_f16<<<NTOK, blk256, 0, stream>>>(ckv32, kv_norm_w, ckvn16, NKVR, NKVR + DROT, NKVR);
    // 7. kv = ckvn @ wkv_up
    gemm_wmma_f16<<<ggrid(NH * 2 * DN, NTOK), blk256, 0, stream>>>(ckvn16, wkuT, nullptr, kv32, NTOK, NH * 2 * DN, NKVR);
    // 8. RoPE + pack q / k / v
    rope_pack_q<<<(NTOK * NH) / 256, blk256, 0, stream>>>(q32, qh16);
    rope_pack_k<<<(NTOK * NH) / 256, blk256, 0, stream>>>(kv32, ckv32, kh16, vT16);
    // 9. flash attention
    attn_wmma<<<dim3(NS / 64, NH, NB), blk128, 0, stream>>>(qh16, kh16, vT16, o16);
    // 10. h = x + o @ wo
    gemm_wmma_f16<<<ggrid(ND, NTOK), blk256, 0, stream>>>(o16, woT, x, h32, NTOK, ND, ND);
    // 11. hf = rmsnorm(h) -> f16
    rmsnorm_f32_to_f16<<<NTOK, blk256, 0, stream>>>(h32, ffn_norm_w, hf16, ND, ND, ND);
    // 12/13. gate & up projections
    gemm_wmma_f16<<<ggrid(NFF, NTOK), blk256, 0, stream>>>(hf16, wgT, nullptr, gate32, NTOK, NFF, ND);
    gemm_wmma_f16<<<ggrid(NFF, NTOK), blk256, 0, stream>>>(hf16, wuT, nullptr, up32, NTOK, NFF, ND);
    // 14. a = silu(gate)*up -> f16
    silu_mul<<<(NTOK * NFF) / 256, blk256, 0, stream>>>(gate32, up32, a16, NTOK * NFF);
    // 15. out = h + a @ w_down
    gemm_wmma_f16<<<ggrid(ND, NTOK), blk256, 0, stream>>>(a16, wdT, h32, out, NTOK, ND, NFF);
}
